// LSTMRemainingSegments_55379308315491
// MI455X (gfx1250) — compile-verified, matrix operand order checked by reading, not run on hardware
//
#include <hip/hip_runtime.h>

// ---------------------------------------------------------------------------
// LSTMRemainingSegments for MI455X (gfx1250, wave32, WMMA 16x16x32 f16).
//
//  * One-time bilinear factorization (batched WMMA GEMMs over o):
//      Abg[b,j,o] = sum_i ctx[b,i]*bg_W[o,i,j]  (f32,  1.5 MB)
//      Abw[b,j,o] = sum_i ctx[b,i]*bw_W[o,i,j]  (f16, 80 MB -> L2-resident)
//    turning the 41 GFLOP/step bilinear into an 80 MB/step L2 stream.
//  * Per step: coalesced bilinear mat-vec, WMMA GEMMs for fc+tanh, fused
//    dual-GEMM LSTM gates (x@Wih.T + h@Whh.T in one kernel), output heads,
//    and elementwise LSTM-cell kernels.
//  * GEMM: 32x32 C macro-tile per wave, 4 v_wmma per 32-K chunk with A/B
//    fragment reuse. Wave id is readfirstlane'd so all tile-selection
//    branches are scalar (EXEC stays all-1s around every v_wmma).
//  * Scratch requirement: ~95 MB in d_ws.
// ---------------------------------------------------------------------------

typedef __attribute__((ext_vector_type(16))) _Float16 v16h;
typedef __attribute__((ext_vector_type(2)))  _Float16 v2h;
typedef __attribute__((ext_vector_type(8)))  float    v8f;

#define B_   256
#define MS_  20
#define DOF_ 6
#define WSZ_ 300
#define CTX_ 512
#define PG_  256
#define PW_  512
#define HG_  256
#define HW_  512
#define JW_  (DOF_ + WSZ_)               // 306
#define GOALS_FLOATS (B_ * MS_ * DOF_)   // 30720

__device__ __forceinline__ int imin(int a, int b) { return a < b ? a : b; }

__device__ __forceinline__ void cvt8(v16h& d, int base, float4 x, float4 y) {
    d[base + 0] = (_Float16)x.x; d[base + 1] = (_Float16)x.y;
    d[base + 2] = (_Float16)x.z; d[base + 3] = (_Float16)x.w;
    d[base + 4] = (_Float16)y.x; d[base + 5] = (_Float16)y.y;
    d[base + 6] = (_Float16)y.z; d[base + 7] = (_Float16)y.w;
}

// Row-major fragment (A row, or B column from an NxK row-major weight):
// lane holds 16 f16: K = k+[0..7] and k+16+[0..7]  (k already includes kb).
__device__ __forceinline__ v16h load_frag_row(const float* __restrict__ base,
                                              int ld, int row, int k) {
    const float* p = base + (size_t)row * ld + k;
    float4 x0 = *reinterpret_cast<const float4*>(p);
    float4 x1 = *reinterpret_cast<const float4*>(p + 4);
    float4 x2 = *reinterpret_cast<const float4*>(p + 16);
    float4 x3 = *reinterpret_cast<const float4*>(p + 20);
    v16h f;
    cvt8(f, 0, x0, x1);
    cvt8(f, 8, x2, x3);
    return f;
}

// K-major fragment (B column from a KxN matrix): strided scalar gather.
__device__ __forceinline__ v16h load_frag_kmaj(const float* __restrict__ base,
                                               int ld, int col, int k) {
    const float* p = base + (size_t)k * ld + col;
    v16h f;
    #pragma unroll
    for (int e = 0; e < 8; ++e) {
        f[e]     = (_Float16)p[(size_t)e * ld];
        f[8 + e] = (_Float16)p[(size_t)(e + 16) * ld];
    }
    return f;
}

__device__ __forceinline__ v8f wmma32(v16h a, v16h b, v8f c) {
    return __builtin_amdgcn_wmma_f32_16x16x32_f16(false, a, false, b,
                                                  (short)0, c, false, false);
}

// Store one 16x16 tile. C/D layout: VGPR r, lanes 0-15 -> M=tm+r,
// lanes 16-31 -> M=tm+r+8; N = tn + (lane&15).
template <int C16>
__device__ __forceinline__ void store_tile(void* __restrict__ Cv, int ldc, int ncs,
                                           const float* __restrict__ bias,
                                           const float* __restrict__ bias2,
                                           int M, int N, int tm, int tn,
                                           int lane, v8f acc, int act) {
    const int l15 = lane & 15;
    const int col = tn + l15;
    if (col >= N) return;
    float bv = bias ? bias[col] : 0.0f;
    if (bias2) bv += bias2[col];
    const int rbase = tm + ((lane >> 4) << 3);
    #pragma unroll
    for (int r = 0; r < 8; ++r) {
        const int row = rbase + r;
        if (row < M) {
            const size_t off = (size_t)row * ldc + (size_t)col * ncs;
            float v = acc[r] + bv;
            if (act) v = tanhf(v);
            if (C16) ((_Float16*)Cv)[off] = (_Float16)v;
            else     ((float*)Cv)[off] = v;
        }
    }
}

// Batched WMMA GEMM, 32x32 macro-tile per wave:
//   C[z][m,n] = act( sum_k A[m,k]*B'(k,n) [+ sum_k A2[m,k]*B2'(k,n)]
//                    + bias[n] [+ bias2[n]] )
//   BT=1: B'(k,n) = Bm[n*ldb + k] (X @ W^T);  BT=0: B'(k,n) = Bm[k*ldb + n].
//   C element (m,n) at C + m*ldc + n*ncs (f32, or f16 when C16=1).
template <int BT, int C16>
__global__ __launch_bounds__(128)
void wmma_gemm_kernel(const float* __restrict__ A, int lda, long long strideA,
                      const float* __restrict__ Bm, int ldb, long long strideB,
                      const float* __restrict__ A2, int lda2,
                      const float* __restrict__ Bm2, int ldb2, int K2,
                      const float* __restrict__ bias,
                      const float* __restrict__ bias2,
                      void* __restrict__ Cv, int ldc, int ncs, long long strideC,
                      int M, int N, int K, int act)
{
    const int lane = threadIdx.x & 31;
    // Force the wave id into an SGPR: all tile-selection logic below becomes
    // provably wave-uniform -> scalar branches, EXEC untouched (WMMA-safe).
    const int wave = __builtin_amdgcn_readfirstlane((int)(threadIdx.x >> 5));
    const int tn0 = (blockIdx.x * 4 + wave) * 32;
    const int tm0 = blockIdx.y * 32;
    if (tn0 >= N || tm0 >= M) return;      // scalar-uniform exit
    const int tn1 = tn0 + 16;
    const int tm1 = tm0 + 16;
    const bool hasN1 = (tn1 < N);          // scalar
    const bool hasM1 = (tm1 < M);          // scalar

    const long long z = blockIdx.z;
    A  += z * strideA;
    Bm += z * strideB;
    char* C = (char*)Cv + z * strideC * (C16 ? sizeof(_Float16) : sizeof(float));

    const int l15 = lane & 15;
    const int kb  = (lane >> 4) << 3;      // 0 (lanes 0-15) or 8 (lanes 16-31)
    const int am0 = imin(tm0 + l15, M - 1);
    const int am1 = imin(tm1 + l15, M - 1);
    const int bn0 = imin(tn0 + l15, N - 1);
    const int bn1 = imin(tn1 + l15, N - 1);

    v8f acc00 = {}, acc01 = {}, acc10 = {}, acc11 = {};

    for (int k0 = 0; k0 < K; k0 += 32) {
        const int k = k0 + kb;
        v16h af0 = load_frag_row(A, lda, am0, k);
        v16h bf0 = BT ? load_frag_row(Bm, ldb, bn0, k)
                      : load_frag_kmaj(Bm, ldb, bn0, k);
        acc00 = wmma32(af0, bf0, acc00);
        if (hasM1) {
            v16h af1 = load_frag_row(A, lda, am1, k);
            acc10 = wmma32(af1, bf0, acc10);
            if (hasN1) {
                v16h bf1 = BT ? load_frag_row(Bm, ldb, bn1, k)
                              : load_frag_kmaj(Bm, ldb, bn1, k);
                acc01 = wmma32(af0, bf1, acc01);
                acc11 = wmma32(af1, bf1, acc11);
            }
        } else if (hasN1) {
            v16h bf1 = BT ? load_frag_row(Bm, ldb, bn1, k)
                          : load_frag_kmaj(Bm, ldb, bn1, k);
            acc01 = wmma32(af0, bf1, acc01);
        }
    }
    if (A2) {  // fused second product (LSTM: + h @ Whh^T), BT layout
        for (int k0 = 0; k0 < K2; k0 += 32) {
            const int k = k0 + kb;
            v16h af0 = load_frag_row(A2, lda2, am0, k);
            v16h bf0 = load_frag_row(Bm2, ldb2, bn0, k);
            acc00 = wmma32(af0, bf0, acc00);
            if (hasM1) {
                v16h af1 = load_frag_row(A2, lda2, am1, k);
                acc10 = wmma32(af1, bf0, acc10);
                if (hasN1) {
                    v16h bf1 = load_frag_row(Bm2, ldb2, bn1, k);
                    acc01 = wmma32(af0, bf1, acc01);
                    acc11 = wmma32(af1, bf1, acc11);
                }
            } else if (hasN1) {
                v16h bf1 = load_frag_row(Bm2, ldb2, bn1, k);
                acc01 = wmma32(af0, bf1, acc01);
            }
        }
    }

    store_tile<C16>(C, ldc, ncs, bias, bias2, M, N, tm0, tn0, lane, acc00, act);
    if (hasN1) store_tile<C16>(C, ldc, ncs, bias, bias2, M, N, tm0, tn1, lane, acc01, act);
    if (hasM1) {
        store_tile<C16>(C, ldc, ncs, bias, bias2, M, N, tm1, tn0, lane, acc10, act);
        if (hasN1) store_tile<C16>(C, ldc, ncs, bias, bias2, M, N, tm1, tn1, lane, acc11, act);
    }
}

// Per-step g-bilinear mat-vec (f32 factor): out[b,o] = bias[o]+sum_j F[b,j,o]*z[b,j]
__global__ void bilinear_f32_kernel(const float* __restrict__ F, int J, int O,
                                    const float* __restrict__ z1, int n1,
                                    const float* __restrict__ bias,
                                    float* __restrict__ out)
{
    __shared__ float zsh[32];
    const int b = blockIdx.x;
    for (int i = threadIdx.x; i < n1; i += blockDim.x) zsh[i] = z1[(size_t)b * n1 + i];
    __syncthreads();
    const int o = threadIdx.x;
    float acc = bias[o];
    const float* fp = F + (size_t)b * J * O + o;
    for (int j = 0; j < J; ++j) acc += fp[(size_t)j * O] * zsh[j];
    out[(size_t)b * O + o] = acc;
}

// Per-step w-bilinear mat-vec (f16 factor, L2-resident stream). Each thread
// handles two adjacent o via packed 2xf16 loads (coalesced).
__global__ void bilinear_f16_kernel(const _Float16* __restrict__ F, int J, int O,
                                    const float* __restrict__ z1, int n1,
                                    const float* __restrict__ z2, int n2,
                                    const float* __restrict__ bias,
                                    float* __restrict__ out)
{
    __shared__ float zsh[320];
    const int b = blockIdx.x;
    for (int i = threadIdx.x; i < n1; i += blockDim.x) zsh[i] = z1[(size_t)b * n1 + i];
    for (int i = threadIdx.x; i < n2; i += blockDim.x) zsh[n1 + i] = z2[(size_t)b * n2 + i];
    __syncthreads();
    const int o2 = threadIdx.x * 2;
    float acc0 = bias[o2], acc1 = bias[o2 + 1];
    const _Float16* fp = F + (size_t)b * J * O + o2;
    for (int j = 0; j < J; ++j) {
        v2h v = *reinterpret_cast<const v2h*>(fp + (size_t)j * O);
        const float zj = zsh[j];
        acc0 += (float)v[0] * zj;
        acc1 += (float)v[1] * zj;
    }
    out[(size_t)b * O + o2]     = acc0;
    out[(size_t)b * O + o2 + 1] = acc1;
}

// LSTM cell nonlinearity (PyTorch gate order i,f,g,o). Grid covers exactly B*H.
__global__ void lstm_pointwise_kernel(const float* __restrict__ g,
                                      float* __restrict__ c,
                                      float* __restrict__ h, int H)
{
    const int idx = blockIdx.x * blockDim.x + threadIdx.x;
    const int b = idx / H, k = idx - b * H;
    const float* gb = g + (size_t)b * 4 * H;
    const float gi = gb[k], gf = gb[H + k], gg = gb[2 * H + k], go = gb[3 * H + k];
    const float si = 1.0f / (1.0f + expf(-gi));
    const float sf = 1.0f / (1.0f + expf(-gf));
    const float so = 1.0f / (1.0f + expf(-go));
    const float cn = sf * c[idx] + si * tanhf(gg);
    c[idx] = cn;
    h[idx] = so * tanhf(cn);
}

// Scatter current (pg, pw) into goals[:, step, :] and ws[:, step, :] of d_out.
__global__ void scatter_kernel(const float* __restrict__ pg,
                               const float* __restrict__ pw,
                               float* __restrict__ out, int step)
{
    const int t = blockIdx.x * blockDim.x + threadIdx.x;
    const int NG = B_ * DOF_;
    const int NW = B_ * WSZ_;
    if (t < NG) {
        const int b = t / DOF_, j = t - b * DOF_;
        out[(size_t)b * MS_ * DOF_ + step * DOF_ + j] = pg[t];
    } else if (t < NG + NW) {
        const int u = t - NG;
        const int b = u / WSZ_, j = u - b * WSZ_;
        out[GOALS_FLOATS + (size_t)b * MS_ * WSZ_ + step * WSZ_ + j] = pw[u];
    }
}

struct GemmCfg {
    const float* A; int lda; long long sA;
    const float* Bm; int ldb; long long sB;
    const float* A2; int lda2; const float* Bm2; int ldb2; int K2;
    const float* bias; const float* bias2;
    void* C; int ldc; int ncs; long long sC;
    int M, N, K, act, Z;
};

template <int BT, int C16>
static inline void launch_gemm(hipStream_t s, const GemmCfg& g)
{
    const int nt32 = (g.N + 31) / 32;
    dim3 grid((nt32 + 3) / 4, (g.M + 31) / 32, g.Z);
    wmma_gemm_kernel<BT, C16><<<grid, dim3(128), 0, s>>>(
        g.A, g.lda, g.sA, g.Bm, g.ldb, g.sB,
        g.A2, g.lda2, g.Bm2, g.ldb2, g.K2,
        g.bias, g.bias2, g.C, g.ldc, g.ncs, g.sC, g.M, g.N, g.K, g.act);
}

extern "C" void kernel_launch(void* const* d_in, const int* in_sizes, int n_in,
                              void* d_out, int out_size, void* d_ws, size_t ws_size,
                              hipStream_t stream)
{
    (void)in_sizes; (void)n_in; (void)out_size; (void)ws_size;

    const float* context = (const float*)d_in[0];
    const float* goal0   = (const float*)d_in[1];
    const float* w0      = (const float*)d_in[2];
    const float* bg_W    = (const float*)d_in[3];
    const float* bg_b    = (const float*)d_in[4];
    const float* bw_W    = (const float*)d_in[5];
    const float* bw_b    = (const float*)d_in[6];
    const float* fcg_W   = (const float*)d_in[7];
    const float* fcg_b   = (const float*)d_in[8];
    const float* fcw_W   = (const float*)d_in[9];
    const float* fcw_b   = (const float*)d_in[10];
    const float* lg_Wih  = (const float*)d_in[11];
    const float* lg_Whh  = (const float*)d_in[12];
    const float* lg_bih  = (const float*)d_in[13];
    const float* lg_bhh  = (const float*)d_in[14];
    const float* lw_Wih  = (const float*)d_in[15];
    const float* lw_Whh  = (const float*)d_in[16];
    const float* lw_bih  = (const float*)d_in[17];
    const float* lw_bhh  = (const float*)d_in[18];
    const float* og_W    = (const float*)d_in[19];
    const float* og_b    = (const float*)d_in[20];
    const float* ow_W    = (const float*)d_in[21];
    const float* ow_b    = (const float*)d_in[22];
    float* out = (float*)d_out;

    // ---- scratch layout (bytes, 256B aligned) ----
    char* wsp = (char*)d_ws;
    size_t off = 0;
    auto alloc = [&](size_t bytes) {
        char* p = wsp + off;
        off += (bytes + 255) & ~(size_t)255;
        return p;
    };
    float*     Abg = (float*)    alloc((size_t)B_ * DOF_ * PG_ * 4);   //  1.5 MB f32
    _Float16*  Abw = (_Float16*) alloc((size_t)B_ * JW_ * PW_ * 2);    // 80.2 MB f16
    float* pg = (float*)alloc((size_t)B_ * DOF_ * 4);
    float* pw = (float*)alloc((size_t)B_ * WSZ_ * 4);
    float* xb = (float*)alloc((size_t)B_ * PG_ * 4);
    float* xt = (float*)alloc((size_t)B_ * PG_ * 4);
    float* yb = (float*)alloc((size_t)B_ * PW_ * 4);
    float* yt = (float*)alloc((size_t)B_ * PW_ * 4);
    float* gG = (float*)alloc((size_t)B_ * 4 * HG_ * 4);
    float* gW = (float*)alloc((size_t)B_ * 4 * HW_ * 4);
    float* gh = (float*)alloc((size_t)2 * B_ * HG_ * 4);
    float* gc = (float*)alloc((size_t)2 * B_ * HG_ * 4);
    float* wh = (float*)alloc((size_t)2 * B_ * HW_ * 4);
    float* wc = (float*)alloc((size_t)2 * B_ * HW_ * 4);

    // ---- init states / first outputs ----
    hipMemsetAsync(gh, 0, (size_t)2 * B_ * HG_ * 4, stream);
    hipMemsetAsync(gc, 0, (size_t)2 * B_ * HG_ * 4, stream);
    hipMemsetAsync(wh, 0, (size_t)2 * B_ * HW_ * 4, stream);
    hipMemsetAsync(wc, 0, (size_t)2 * B_ * HW_ * 4, stream);
    hipMemcpyAsync(pg, goal0, (size_t)B_ * DOF_ * 4, hipMemcpyDeviceToDevice, stream);
    hipMemcpyAsync(pw, w0, (size_t)B_ * WSZ_ * 4, hipMemcpyDeviceToDevice, stream);

    const int scat_grid = (B_ * (DOF_ + WSZ_) + 255) / 256;
    scatter_kernel<<<scat_grid, 256, 0, stream>>>(pg, pw, out, 0);

    // ---- one-time bilinear factorization (batched over o) ----
    {   // Abg[b,j,o] (f32): per-o GEMM context(256x512) @ bg_W[o](512x6 K-major)
        GemmCfg g = { context, CTX_, 0, bg_W, DOF_, (long long)CTX_ * DOF_,
                      nullptr, 0, nullptr, 0, 0, nullptr, nullptr,
                      Abg, DOF_ * PG_, PG_, 1, B_, DOF_, CTX_, 0, PG_ };
        launch_gemm<0, 0>(stream, g);
    }
    {   // Abw[b,j,o] (f16): per-o GEMM context(256x512) @ bw_W[o](512x306 K-major)
        GemmCfg g = { context, CTX_, 0, bw_W, JW_, (long long)CTX_ * JW_,
                      nullptr, 0, nullptr, 0, 0, nullptr, nullptr,
                      Abw, JW_ * PW_, PW_, 1, B_, JW_, CTX_, 0, PW_ };
        launch_gemm<0, 1>(stream, g);
    }

    // ---- 19 sequential segments ----
    for (int s = 1; s < MS_; ++s) {
        bilinear_f32_kernel<<<B_, PG_, 0, stream>>>(Abg, DOF_, PG_, pg, DOF_, bg_b, xb);
        bilinear_f16_kernel<<<B_, PW_ / 2, 0, stream>>>(Abw, JW_, PW_, pg, DOF_,
                                                        pw, WSZ_, bw_b, yb);
        {   // xt = tanh(xb @ fcg_W.T + fcg_b)
            GemmCfg g = { xb, PG_, 0, fcg_W, PG_, 0, nullptr, 0, nullptr, 0, 0,
                          fcg_b, nullptr, xt, PG_, 1, 0, B_, PG_, PG_, 1, 1 };
            launch_gemm<1, 0>(stream, g);
        }
        {   // yt = tanh(yb @ fcw_W.T + fcw_b)
            GemmCfg g = { yb, PW_, 0, fcw_W, PW_, 0, nullptr, 0, nullptr, 0, 0,
                          fcw_b, nullptr, yt, PW_, 1, 0, B_, PW_, PW_, 1, 1 };
            launch_gemm<1, 0>(stream, g);
        }
        // goal LSTM (2 layers, H=256): fused  inp@Wih.T + h@Whh.T + bih + bhh
        for (int l = 0; l < 2; ++l) {
            const float* inp = (l == 0) ? xt : gh;
            float* hl = gh + (size_t)l * B_ * HG_;
            float* cl = gc + (size_t)l * B_ * HG_;
            GemmCfg g = { inp, HG_, 0,
                          lg_Wih + (size_t)l * 4 * HG_ * PG_, PG_, 0,
                          hl, HG_, lg_Whh + (size_t)l * 4 * HG_ * HG_, HG_, HG_,
                          lg_bih + (size_t)l * 4 * HG_, lg_bhh + (size_t)l * 4 * HG_,
                          gG, 4 * HG_, 1, 0, B_, 4 * HG_, PG_, 0, 1 };
            launch_gemm<1, 0>(stream, g);
            lstm_pointwise_kernel<<<(B_ * HG_) / 256, 256, 0, stream>>>(gG, cl, hl, HG_);
        }
        // waypoint LSTM (2 layers, H=512)
        for (int l = 0; l < 2; ++l) {
            const float* inp = (l == 0) ? yt : wh;
            float* hl = wh + (size_t)l * B_ * HW_;
            float* cl = wc + (size_t)l * B_ * HW_;
            GemmCfg g = { inp, HW_, 0,
                          lw_Wih + (size_t)l * 4 * HW_ * PW_, PW_, 0,
                          hl, HW_, lw_Whh + (size_t)l * 4 * HW_ * HW_, HW_, HW_,
                          lw_bih + (size_t)l * 4 * HW_, lw_bhh + (size_t)l * 4 * HW_,
                          gW, 4 * HW_, 1, 0, B_, 4 * HW_, PW_, 0, 1 };
            launch_gemm<1, 0>(stream, g);
            lstm_pointwise_kernel<<<(B_ * HW_) / 256, 256, 0, stream>>>(gW, cl, hl, HW_);
        }
        {   // pg = gh[1] @ og_W.T + og_b
            GemmCfg g = { gh + (size_t)B_ * HG_, HG_, 0, og_W, HG_, 0,
                          nullptr, 0, nullptr, 0, 0, og_b, nullptr,
                          pg, DOF_, 1, 0, B_, DOF_, HG_, 0, 1 };
            launch_gemm<1, 0>(stream, g);
        }
        {   // pw = wh[1] @ ow_W.T + ow_b
            GemmCfg g = { wh + (size_t)B_ * HW_, HW_, 0, ow_W, HW_, 0,
                          nullptr, 0, nullptr, 0, 0, ow_b, nullptr,
                          pw, WSZ_, 1, 0, B_, WSZ_, HW_, 0, 1 };
            launch_gemm<1, 0>(stream, g);
        }
        scatter_kernel<<<scat_grid, 256, 0, stream>>>(pg, pw, out, s);
    }
}
